// MambaBlocks_84413287235934
// MI455X (gfx1250) — compile-verified, matrix operand order checked by reading, not run on hardware
//
#include <hip/hip_runtime.h>
#include <hip/hip_bf16.h>

// ---------------- problem constants ----------------
#define B_  4
#define L_  1024
#define D_  768
#define DI_ 1536
#define N_  16
#define K_  4
#define R_  48
#define RPAD_ 64         // R padded to WMMA K multiple
#define RN2_ 80          // R + 2N
#define DEPTH_ 24
#define M_  (B_ * L_)    // 4096
#define EPS_ 1e-5f

// ---------------- WMMA types ----------------
typedef __attribute__((ext_vector_type(16))) __bf16 v16bf;
typedef __attribute__((ext_vector_type(8)))  float  v8f;

union FragBF { v16bf v; unsigned int u[8]; };

__device__ __forceinline__ unsigned short f32_to_bf16(float f) {
    unsigned int u = __float_as_uint(f);
    unsigned int r = (u + 0x7FFFu + ((u >> 16) & 1u)) >> 16;
    return (unsigned short)r;
}

__device__ __forceinline__ float silu(float x) {
    return x / (1.f + __expf(-x));
}

// ---------------- bf16 WMMA GEMM: C[M,N] = A[M,K] * B[N,K]^T ----------------
// Guard-free fast path:
//   * M  must be a multiple of BM (always 4096 here)
//   * Kk must be a multiple of BK (dtproj is pre-padded 48 -> 64)
//   * B rows may be over-read up to the next BN multiple; caller guarantees
//     the backing buffer is large enough (wbf scratch). Only the C store
//     checks col < Nn.
#define BM 128
#define BN 128
#define BK 32
#define LDSPAD 8

__global__ __launch_bounds__(256) void k_gemm_bf16(
    const unsigned short* __restrict__ A, int lda,
    const unsigned short* __restrict__ Bw, int ldb,
    float* __restrict__ C, int ldc,
    int Mm, int Nn, int Kk)
{
    __shared__ unsigned short As[BM][BK + LDSPAD];
    __shared__ unsigned short Bs[BN][BK + LDSPAD];

    const int tid  = threadIdx.x;
    const int wid  = tid >> 5;
    const int lane = tid & 31;
    const int lrow = lane & 15;
    const int g    = lane >> 4;      // K-half select per WMMA 16-bit layout
    const int wm   = wid >> 1;       // 0..3  (wave tile row: 32 rows)
    const int wn   = wid & 1;        // 0..1  (wave tile col: 64 cols)
    const int tileM = blockIdx.y * BM;
    const int tileN = blockIdx.x * BN;

    v8f acc[2][4];
    for (int i = 0; i < 2; ++i)
        for (int j = 0; j < 4; ++j)
            for (int e = 0; e < 8; ++e) acc[i][j][e] = 0.f;

    const int srow = tid >> 1;             // 0..127
    const int scb  = (tid & 1) * 16;       // 0 or 16

    for (int k0 = 0; k0 < Kk; k0 += BK) {
        // --- stage A tile 128x32 and B tile 128x32 (vector, unguarded) ---
        {
            const unsigned short* ap = A + (size_t)(tileM + srow) * lda + k0 + scb;
            uint4* dst = (uint4*)&As[srow][scb];
            dst[0] = ((const uint4*)ap)[0];
            dst[1] = ((const uint4*)ap)[1];
        }
        {
            const unsigned short* bp = Bw + (size_t)(tileN + srow) * ldb + k0 + scb;
            uint4* dst = (uint4*)&Bs[srow][scb];
            dst[0] = ((const uint4*)bp)[0];
            dst[1] = ((const uint4*)bp)[1];
        }
        __syncthreads();

        // --- fragment gathers per CDNA5 16-bit WMMA layout ---
        FragBF a[2], b[4];
        #pragma unroll
        for (int i = 0; i < 2; ++i) {
            int m = wm * 32 + i * 16 + lrow;
            const unsigned short* src = &As[m][0];
            #pragma unroll
            for (int e2 = 0; e2 < 8; ++e2) {
                int k = (e2 < 4 ? 2 * e2 : 2 * e2 + 8) + 8 * g;
                a[i].u[e2] = *(const unsigned int*)&src[k];
            }
        }
        #pragma unroll
        for (int j = 0; j < 4; ++j) {
            int n = wn * 64 + j * 16 + lrow;
            const unsigned short* src = &Bs[n][0];
            #pragma unroll
            for (int e2 = 0; e2 < 8; ++e2) {
                int k = (e2 < 4 ? 2 * e2 : 2 * e2 + 8) + 8 * g;
                b[j].u[e2] = *(const unsigned int*)&src[k];
            }
        }
        #pragma unroll
        for (int i = 0; i < 2; ++i)
            #pragma unroll
            for (int j = 0; j < 4; ++j)
                acc[i][j] = __builtin_amdgcn_wmma_f32_16x16x32_bf16(
                    false, a[i].v, false, b[j].v,
                    (short)0, acc[i][j], false, false);
        __syncthreads();
    }

    // --- store: VGPR r -> row r + 8*g, col = lane&15 ---
    #pragma unroll
    for (int i = 0; i < 2; ++i)
        #pragma unroll
        for (int j = 0; j < 4; ++j) {
            int col = tileN + wn * 64 + j * 16 + lrow;
            if (col < Nn) {
                #pragma unroll
                for (int r = 0; r < 8; ++r) {
                    int row = tileM + wm * 32 + i * 16 + r + 8 * g;
                    C[(size_t)row * ldc + col] = acc[i][j][r];
                }
            }
        }
}

// ---------------- init: hid = x, res = 0 ----------------
__global__ __launch_bounds__(256) void k_init(const float* __restrict__ x,
                                              float* __restrict__ hid,
                                              float* __restrict__ res, long n)
{
    long i = (long)blockIdx.x * 256 + threadIdx.x;
    if (i < n) { hid[i] = x[i]; res[i] = 0.f; }
}

// ---------------- residual add + RMSNorm -> bf16 ----------------
__global__ __launch_bounds__(256) void k_addnorm(const float* __restrict__ hid,
                                                 float* __restrict__ res,
                                                 const float* __restrict__ w,
                                                 unsigned short* __restrict__ hbf)
{
    int row = blockIdx.x;
    __shared__ float red[8];
    float vals[3];
    float ss = 0.f;
    #pragma unroll
    for (int i = 0; i < 3; ++i) {
        int c = threadIdx.x + i * 256;
        float v = res[(size_t)row * D_ + c] + hid[(size_t)row * D_ + c];
        vals[i] = v; ss += v * v;
    }
    for (int off = 16; off; off >>= 1) ss += __shfl_xor(ss, off, 32);
    if ((threadIdx.x & 31) == 0) red[threadIdx.x >> 5] = ss;
    __syncthreads();
    if (threadIdx.x == 0) {
        float t = 0.f;
        for (int i = 0; i < 8; ++i) t += red[i];
        red[0] = t;
    }
    __syncthreads();
    float scale = rsqrtf(red[0] / (float)D_ + EPS_);
    #pragma unroll
    for (int i = 0; i < 3; ++i) {
        int c = threadIdx.x + i * 256;
        float v = vals[i];
        res[(size_t)row * D_ + c] = v;
        hbf[(size_t)row * D_ + c] = f32_to_bf16(v * scale * w[c]);
    }
}

// ---------------- final RMSNorm -> f32 out ----------------
__global__ __launch_bounds__(256) void k_final(const float* __restrict__ hid,
                                               const float* __restrict__ res,
                                               const float* __restrict__ w,
                                               float* __restrict__ out)
{
    int row = blockIdx.x;
    __shared__ float red[8];
    float vals[3];
    float ss = 0.f;
    #pragma unroll
    for (int i = 0; i < 3; ++i) {
        int c = threadIdx.x + i * 256;
        float v = res[(size_t)row * D_ + c] + hid[(size_t)row * D_ + c];
        vals[i] = v; ss += v * v;
    }
    for (int off = 16; off; off >>= 1) ss += __shfl_xor(ss, off, 32);
    if ((threadIdx.x & 31) == 0) red[threadIdx.x >> 5] = ss;
    __syncthreads();
    if (threadIdx.x == 0) {
        float t = 0.f;
        for (int i = 0; i < 8; ++i) t += red[i];
        red[0] = t;
    }
    __syncthreads();
    float scale = rsqrtf(red[0] / (float)D_ + EPS_);
    #pragma unroll
    for (int i = 0; i < 3; ++i) {
        int c = threadIdx.x + i * 256;
        out[(size_t)row * D_ + c] = vals[i] * scale * w[c];
    }
}

// ---------------- f32 -> bf16 conversions ----------------
__global__ __launch_bounds__(256) void k_f2bf_flat(const float* __restrict__ src,
                                                   unsigned short* __restrict__ dst, long n)
{
    long i = (long)blockIdx.x * 256 + threadIdx.x;
    if (i < n) dst[i] = f32_to_bf16(src[i]);
}

// convert with column padding: dst[r*ldd + c] = c<cols ? bf16(src[r*lds+c]) : 0
__global__ __launch_bounds__(256) void k_f2bf_pad(const float* __restrict__ src, int lds_,
                                                  unsigned short* __restrict__ dst, int ldd,
                                                  int cols, long rows)
{
    long i = (long)blockIdx.x * 256 + threadIdx.x;
    if (i >= rows * (long)ldd) return;
    long r = i / ldd;
    int  c = (int)(i - r * ldd);
    dst[i] = (c < cols) ? f32_to_bf16(src[r * (long)lds_ + c]) : (unsigned short)0;
}

// ---------------- depthwise causal conv (K=4) + SiLU ----------------
__global__ __launch_bounds__(256) void k_conv_silu(const float* __restrict__ uraw, int ldu,
                                                   const float* __restrict__ cw,
                                                   const float* __restrict__ cb,
                                                   float* __restrict__ uc,
                                                   unsigned short* __restrict__ ucbf,
                                                   int reverse)
{
    long i = (long)blockIdx.x * 256 + threadIdx.x;
    if (i >= (long)B_ * L_ * DI_) return;
    int  d  = (int)(i % DI_);
    long bl = i / DI_;
    int  tt = (int)(bl % L_);
    int  b  = (int)(bl / L_);
    float s = cb[d];
    #pragma unroll
    for (int j = 0; j < K_; ++j) {
        int sidx = tt - (K_ - 1) + j;
        if (sidx >= 0) {
            int tp = reverse ? (L_ - 1 - sidx) : sidx;
            s += cw[d * K_ + j] * uraw[((size_t)b * L_ + tp) * ldu + d];
        }
    }
    float v = silu(s);
    uc[i]   = v;
    ucbf[i] = f32_to_bf16(v);
}

// ---------------- softplus(dt + bias) in-place ----------------
__global__ __launch_bounds__(256) void k_softplus(float* __restrict__ dt,
                                                  const float* __restrict__ db, long n)
{
    long i = (long)blockIdx.x * 256 + threadIdx.x;
    if (i >= n) return;
    int d = (int)(i % DI_);
    float x = dt[i] + db[d];
    dt[i] = (x > 20.f) ? x : log1pf(__expf(x));
}

// ---------------- selective scan + gate ----------------
// block = 256 channels of one batch. B/C rows staged in LDS in 64-step chunks
// (8 KB) with float4 loads; per-step LDS reads are wave broadcasts.
#define SCHUNK 64

__global__ __launch_bounds__(256) void k_scan(const float* __restrict__ uc,
                                              const float* __restrict__ dt,
                                              const float* __restrict__ dbl_,
                                              const float* __restrict__ Alog,
                                              const float* __restrict__ Dv,
                                              const float* __restrict__ z, int ldz,
                                              float* __restrict__ y,
                                              int reverse, int accumulate)
{
    const int nchunk = DI_ / 256;
    int b = blockIdx.x / nchunk;
    int d = (blockIdx.x % nchunk) * 256 + threadIdx.x;

    __shared__ float sBC[SCHUNK * 32];   // [tloc][n]: n<16 -> B, n>=16 -> C

    float A[N_], h[N_];
    #pragma unroll
    for (int n = 0; n < N_; ++n) {
        A[n] = -__expf(Alog[(size_t)d * N_ + n]);
        h[n] = 0.f;
    }
    float dskip = Dv[d];

    const int stl = threadIdx.x >> 2;          // staged timestep 0..63
    const int sn0 = (threadIdx.x & 3) * 8;     // staged column base

    for (int c0 = 0; c0 < L_; c0 += SCHUNK) {
        __syncthreads();   // previous chunk fully consumed
        {
            const float* src = &dbl_[((size_t)b * L_ + c0 + stl) * RN2_ + R_ + sn0];
            float4* dst = (float4*)&sBC[stl * 32 + sn0];
            dst[0] = ((const float4*)src)[0];
            dst[1] = ((const float4*)src)[1];
        }
        __syncthreads();

        for (int tl = 0; tl < SCHUNK; ++tl) {
            int tt = c0 + tl;
            size_t m = (size_t)b * L_ + tt;
            const float* bc = &sBC[tl * 32];

            float u_t  = uc[m * DI_ + d];
            float dt_t = dt[m * DI_ + d];
            float du   = dt_t * u_t;
            float yv   = 0.f;
            #pragma unroll
            for (int n = 0; n < N_; ++n) {
                float dA = __expf(dt_t * A[n]);
                h[n] = dA * h[n] + du * bc[n];
                yv  += h[n] * bc[N_ + n];
            }
            yv += u_t * dskip;

            int tp = reverse ? (L_ - 1 - tt) : tt;
            size_t mp = (size_t)b * L_ + tp;
            float zt = z[mp * ldz + d];
            yv *= silu(zt);
            if (accumulate) y[mp * DI_ + d] += yv;
            else            y[mp * DI_ + d]  = yv;
        }
    }
}

// ============================ host driver ============================
static inline long cdiv(long a, long b) { return (a + b - 1) / b; }

extern "C" void kernel_launch(void* const* d_in, const int* in_sizes, int n_in,
                              void* d_out, int out_size, void* d_ws, size_t ws_size,
                              hipStream_t stream)
{
    (void)in_sizes; (void)n_in; (void)out_size; (void)ws_size;

    const float* x         = (const float*)d_in[0];
    const float* norm_w    = (const float*)d_in[1];
    const float* in_proj_w = (const float*)d_in[2];
    const float* conv_w    = (const float*)d_in[3];
    const float* conv_b    = (const float*)d_in[4];
    const float* xproj_w   = (const float*)d_in[5];
    const float* dtproj_w  = (const float*)d_in[6];
    const float* dtproj_b  = (const float*)d_in[7];
    const float* A_log     = (const float*)d_in[8];
    const float* D_skip    = (const float*)d_in[9];
    const float* conv_w_b  = (const float*)d_in[10];
    const float* conv_b_b  = (const float*)d_in[11];
    const float* xproj_w_b = (const float*)d_in[12];
    const float* dtproj_w_b= (const float*)d_in[13];
    const float* dtproj_b_b= (const float*)d_in[14];
    const float* A_b_log   = (const float*)d_in[15];
    const float* D_skip_b  = (const float*)d_in[16];
    const float* out_proj_w= (const float*)d_in[17];
    const float* norm_f_w  = (const float*)d_in[18];

    // ---- workspace carve ----
    char* p = (char*)d_ws;
    auto carve = [&](size_t bytes) {
        void* r = (void*)p;
        p += (bytes + 255) & ~(size_t)255;
        return r;
    };
    float* res   = (float*)carve((size_t)M_ * D_  * 4);
    float* hid   = (float*)carve((size_t)M_ * D_  * 4);
    float* xz    = (float*)carve((size_t)M_ * 2 * DI_ * 4);
    float* uc    = (float*)carve((size_t)M_ * DI_ * 4);
    float* dtb   = (float*)carve((size_t)M_ * DI_ * 4);
    float* ybuf  = (float*)carve((size_t)M_ * DI_ * 4);
    float* dbl   = (float*)carve((size_t)M_ * RN2_ * 4);
    unsigned short* hbf   = (unsigned short*)carve((size_t)M_ * D_  * 2);
    unsigned short* ucbf  = (unsigned short*)carve((size_t)M_ * DI_ * 2);
    unsigned short* dtrbf = (unsigned short*)carve((size_t)M_ * RPAD_ * 2);
    unsigned short* ybf   = (unsigned short*)carve((size_t)M_ * DI_ * 2);
    unsigned short* wbf   = (unsigned short*)carve((size_t)2 * DI_ * D_ * 2);

    const long nMD  = (long)M_ * D_;
    const long nMDI = (long)M_ * DI_;

    k_init<<<(unsigned)cdiv(nMD, 256), 256, 0, stream>>>(x, hid, res, nMD);

    for (int l = 0; l < DEPTH_; ++l) {
        const float* nw   = norm_w     + (size_t)l * D_;
        const float* ipw  = in_proj_w  + (size_t)l * 2 * DI_ * D_;
        const float* opw  = out_proj_w + (size_t)l * D_ * DI_;

        // res += hid; h = rms(res)*w -> hbf
        k_addnorm<<<M_, 256, 0, stream>>>(hid, res, nw, hbf);

        // xz = h @ in_proj_w^T   (M x 3072, K=768)
        k_f2bf_flat<<<(unsigned)cdiv((long)2 * DI_ * D_, 256), 256, 0, stream>>>(
            ipw, wbf, (long)2 * DI_ * D_);
        k_gemm_bf16<<<dim3(2 * DI_ / BN, M_ / BM), 256, 0, stream>>>(
            hbf, D_, wbf, D_, xz, 2 * DI_, M_, 2 * DI_, D_);

        for (int dir = 0; dir < 2; ++dir) {
            const float* cw = dir ? conv_w_b   + (size_t)l * DI_ * K_   : conv_w   + (size_t)l * DI_ * K_;
            const float* cb = dir ? conv_b_b   + (size_t)l * DI_        : conv_b   + (size_t)l * DI_;
            const float* xw = dir ? xproj_w_b  + (size_t)l * RN2_ * DI_ : xproj_w  + (size_t)l * RN2_ * DI_;
            const float* dw = dir ? dtproj_w_b + (size_t)l * DI_ * R_   : dtproj_w + (size_t)l * DI_ * R_;
            const float* db = dir ? dtproj_b_b + (size_t)l * DI_        : dtproj_b + (size_t)l * DI_;
            const float* Al = dir ? A_b_log    + (size_t)l * DI_ * N_   : A_log    + (size_t)l * DI_ * N_;
            const float* Dv = dir ? D_skip_b   + (size_t)l * DI_        : D_skip   + (size_t)l * DI_;

            // uc = silu(causal_conv(u)), in scan-time order (reversed for dir=1)
            k_conv_silu<<<(unsigned)cdiv(nMDI, 256), 256, 0, stream>>>(
                xz /*u view*/, 2 * DI_, cw, cb, uc, ucbf, dir);

            // dbl = uc @ xproj_w^T  (M x 80, K=1536); N tile over-reads wbf rows 80..127
            k_f2bf_flat<<<(unsigned)cdiv((long)RN2_ * DI_, 256), 256, 0, stream>>>(
                xw, wbf, (long)RN2_ * DI_);
            k_gemm_bf16<<<dim3(1, M_ / BM), 256, 0, stream>>>(
                ucbf, DI_, wbf, DI_, dbl, RN2_, M_, RN2_, DI_);

            // dtr (first R cols of dbl) -> bf16, zero-padded to K=64
            k_f2bf_pad<<<(unsigned)cdiv((long)M_ * RPAD_, 256), 256, 0, stream>>>(
                dbl, RN2_, dtrbf, RPAD_, R_, (long)M_);

            // dt_raw = dtr @ dtproj_w^T (M x 1536, K=64 padded)
            k_f2bf_pad<<<(unsigned)cdiv((long)DI_ * RPAD_, 256), 256, 0, stream>>>(
                dw, R_, wbf, RPAD_, R_, (long)DI_);
            k_gemm_bf16<<<dim3(DI_ / BN, M_ / BM), 256, 0, stream>>>(
                dtrbf, RPAD_, wbf, RPAD_, dtb, DI_, M_, DI_, RPAD_);

            // dt = softplus(dt_raw + bias)
            k_softplus<<<(unsigned)cdiv(nMDI, 256), 256, 0, stream>>>(dtb, db, nMDI);

            // selective scan + silu(z) gate; dir==1 accumulates into y
            k_scan<<<B_ * (DI_ / 256), 256, 0, stream>>>(
                uc, dtb, dbl, Al, Dv, xz + DI_ /*z view*/, 2 * DI_, ybuf, dir, dir);
        }

        // hid_next = y @ out_proj_w^T  (M x 768, K=1536)
        k_f2bf_flat<<<(unsigned)cdiv(nMDI, 256), 256, 0, stream>>>(ybuf, ybf, nMDI);
        k_f2bf_flat<<<(unsigned)cdiv((long)D_ * DI_, 256), 256, 0, stream>>>(
            opw, wbf, (long)D_ * DI_);
        k_gemm_bf16<<<dim3(D_ / BN, M_ / BM), 256, 0, stream>>>(
            ybf, DI_, wbf, DI_, hid, D_, M_, D_, DI_);
    }

    k_final<<<M_, 256, 0, stream>>>(hid, res, norm_f_w, (float*)d_out);
}
